// SimpleGCN_49426483642633
// MI455X (gfx1250) — matrix operand order repaired and checked by reading, not compiled, not run on hardware
//
#include <hip/hip_runtime.h>
#include <hip/hip_bf16.h>

typedef __attribute__((ext_vector_type(8)))  float        v8f;
typedef __attribute__((ext_vector_type(16))) __bf16       v16bf;
typedef __attribute__((ext_vector_type(8)))  __bf16       v8bf;
typedef __attribute__((ext_vector_type(4)))  unsigned int v4u;
typedef __attribute__((ext_vector_type(8)))  int          v8i;
typedef __attribute__((ext_vector_type(4)))  int          v4i;

#define N_ROWS 8192
#define NFEAT  512
#define NHID   256
#define NCLASS 128

__device__ __forceinline__ v16bf cat8(v8bf lo, v8bf hi) {
  return __builtin_shufflevector(lo, hi, 0,1,2,3,4,5,6,7,8,9,10,11,12,13,14,15);
}

// A fragment (16 x 32 bf16) per CDNA5 ISA layout:
// lanes 0-15: M=lane,   K = k0+[0..7]  then k0+[16..23]
// lanes16-31: M=lane-16,K = k0+[8..15] then k0+[24..31]
__device__ __forceinline__ v16bf load_frag_a(const __bf16* src, int ld, int row0, int k0) {
  int lane = threadIdx.x & 31;
  int m = lane & 15, hi = lane >> 4;
  const __bf16* p = src + (size_t)(row0 + m) * ld + k0 + hi * 8;
  v8bf c0 = *(const v8bf*)p;
  v8bf c1 = *(const v8bf*)(p + 16);
  return cat8(c0, c1);
}

// B fragment (32 x 16 bf16), B[k][n] = src[n0+n][k0+k] (src row-major, "W^T trick"):
// lanes 0-15: N=lane, K=k0+[0..7] then k0+[16..23]; lanes 16-31: N=lane-16, K=k0+[8..15] then k0+[24..31]
__device__ __forceinline__ v16bf load_frag_b(const __bf16* src, int ld, int n0, int k0) {
  int lane = threadIdx.x & 31;
  int n = lane & 15, hi = lane >> 4;
  const __bf16* p = src + (size_t)(n0 + n) * ld + k0 + hi * 8;
  v8bf c0 = *(const v8bf*)p;
  v8bf c1 = *(const v8bf*)(p + 16);
  return cat8(c0, c1);
}

__device__ __forceinline__ v8f wmma_bf16(v16bf a, v16bf b, v8f c) {
  return __builtin_amdgcn_wmma_f32_16x16x32_bf16(false, a, false, b, (short)0, c, false, false);
}

// 16x16 S tile: rows arow0.. of qs (LDS) x rows jrow0.. of kt (LDS), K = NFEAT.
__device__ __forceinline__ v8f s_tile(const __bf16* qs, int arow0, const __bf16* kt, int jrow0) {
  v8f s = {0.f,0.f,0.f,0.f,0.f,0.f,0.f,0.f};
#pragma unroll
  for (int k = 0; k < NFEAT; k += 32) {
    v16bf a = load_frag_a(qs, NFEAT, arow0, k);
    v16bf b = load_frag_b(kt, NFEAT, jrow0, k);
    s = wmma_bf16(a, b, s);
  }
  return s;
}

// ---- Tensor Data Mover: 2-D tile load, 2-byte elements (D# per CDNA5 ISA 8.3-8.5) ----
// Tile of tile_d1 rows x tile_d0 elems from a tensor with row length tensor_d0,
// row count tensor_d1, row stride stride0 (elements), into LDS at lds_off (bytes).
__device__ __forceinline__ void tdm_load_2d(unsigned lds_off, const void* gbase,
                                            unsigned tensor_d0, unsigned tensor_d1,
                                            unsigned tile_d0, unsigned tile_d1,
                                            unsigned stride0) {
  unsigned long long ga = (unsigned long long)(size_t)gbase;
  v4u g0 = { 1u,                                           // count=1 (valid user D#)
             lds_off,                                      // lds_addr (bytes)
             (unsigned)ga,                                 // global_addr[31:0]
             (unsigned)((ga >> 32) & 0x01FFFFFFu) | (2u << 30) }; // addr[56:32], type=2
  v8i g1 = { (int)(1u << 16),                              // wg_mask=0, data_size=1 (2B)
             (int)((tensor_d0 & 0xFFFFu) << 16),           // tensor_dim0[15:0]
             (int)(((tensor_d0 >> 16) & 0xFFFFu) | ((tensor_d1 & 0xFFFFu) << 16)),
             (int)(((tensor_d1 >> 16) & 0xFFFFu) | ((tile_d0 & 0xFFFFu) << 16)),
             (int)(tile_d1 & 0xFFFFu),                     // tile_dim1, tile_dim2=0
             (int)stride0,                                 // tensor_dim0_stride[31:0]
             0, 0 };
  v4i gz4 = { 0, 0, 0, 0 };
  v8i gz8 = { 0, 0, 0, 0, 0, 0, 0, 0 };
  // clang-23 / therock signature: (v4u, v8i, v4i, v4i, v8i, i32 cpol)
  __builtin_amdgcn_tensor_load_to_lds(g0, g1, gz4, gz4, gz8, 0);
}

__device__ __forceinline__ unsigned lds_offset_of(const void* p) {
  // LDS-aperture generic pointers carry the LDS byte offset in their low 32 bits.
  return (unsigned)(size_t)p;
}

// ---------------- helper kernels ----------------

__global__ void cvt_f32_bf16(const float* __restrict__ in, __bf16* __restrict__ out, int n) {
  int i = blockIdx.x * blockDim.x + threadIdx.x;
  if (i < n) out[i] = (__bf16)in[i];
}

// out[C][R] (bf16) = transpose of in[R][C] (f32); R, C multiples of 32. block 32x8.
__global__ void transpose_f32_bf16(const float* __restrict__ in, __bf16* __restrict__ out,
                                   int R, int C) {
  __shared__ float tile[32][33];
  int c0 = blockIdx.x * 32, r0 = blockIdx.y * 32;
#pragma unroll
  for (int i = threadIdx.y; i < 32; i += 8)
    tile[i][threadIdx.x] = in[(size_t)(r0 + i) * C + c0 + threadIdx.x];
  __syncthreads();
#pragma unroll
  for (int i = threadIdx.y; i < 32; i += 8)
    out[(size_t)(c0 + i) * R + r0 + threadIdx.x] = (__bf16)tile[threadIdx.x][i];
}

// W12[c][k] = sum_h w2[c][h] * w1[h][k]  (128x512, f32 math, bf16 out). tiny.
__global__ void w12_kernel(const float* __restrict__ w1, const float* __restrict__ w2,
                           __bf16* __restrict__ W12) {
  int idx = blockIdx.x * blockDim.x + threadIdx.x;   // 128*512 threads
  int c = idx >> 9, k = idx & 511;
  float s = 0.f;
  for (int h = 0; h < NHID; ++h) s += w2[c * NHID + h] * w1[h * NFEAT + k];
  W12[idx] = (__bf16)s;
}

// ---------------- row statistics: m_i = max_j S_ij, Z_i = sum_j exp(S_ij - m_i) ----------------
// 256 threads = 8 waves, 32 rows per workgroup; K tiles double-buffered via TDM.

__global__ void __launch_bounds__(256) gcn_stats(const __bf16* __restrict__ Qb,
                                                 const __bf16* __restrict__ Ktb,
                                                 float* __restrict__ rowMax,
                                                 float* __restrict__ rowInvZ) {
  __shared__ alignas(16) __bf16 qs[32 * NFEAT];       // 32 KiB
  __shared__ alignas(16) __bf16 kts[2][64 * NFEAT];   // 128 KiB (TDM, double-buffered)
  __shared__ float ss[32 * 64];                       // 8 KiB
  __shared__ float mrow[32], zrow[32];
  int tid = threadIdx.x;
  int rowBase = blockIdx.x * 32;
  int wave = tid >> 5, lane = tid & 31, hi = lane >> 4, nn = lane & 15;
  int rg = wave >> 2, jt = wave & 3;   // row-group, j-tile

  if (wave == 0)                       // prefetch block 0 (overlaps with qs staging)
    tdm_load_2d(lds_offset_of(kts[0]), Ktb, NFEAT, N_ROWS, NFEAT, 64, NFEAT);
  {
    const v8bf* src = (const v8bf*)(Qb + (size_t)rowBase * NFEAT);
    v8bf* dst = (v8bf*)qs;
    for (int i = tid; i < 32 * NFEAT / 8; i += 256) dst[i] = src[i];
  }
  if (tid < 32) { mrow[tid] = -3.0e38f; zrow[tid] = 0.0f; }

  int p = 0;
  for (int jb = 0; jb < N_ROWS; jb += 64, p ^= 1) {
    __syncthreads();                   // buffer p^1 fully consumed; ok to overwrite
    if (wave == 0) {
      if (jb + 64 < N_ROWS) {
        tdm_load_2d(lds_offset_of(kts[p ^ 1]), Ktb + (size_t)(jb + 64) * NFEAT,
                    NFEAT, N_ROWS, NFEAT, 64, NFEAT);
        __builtin_amdgcn_s_wait_tensorcnt(1);   // block jb done; prefetch in flight
      } else {
        __builtin_amdgcn_s_wait_tensorcnt(0);
      }
    }
    __syncthreads();
    v8f s = s_tile(qs, rg * 16, kts[p], jt * 16);
#pragma unroll
    for (int r = 0; r < 8; ++r)
      ss[(rg * 16 + r + hi * 8) * 64 + jt * 16 + nn] = s[r];
    __syncthreads();
    if (tid < 32) {
      float m0 = mrow[tid];
      float bm = m0;
      for (int c = 0; c < 64; ++c) bm = fmaxf(bm, ss[tid * 64 + c]);
      float z = zrow[tid] * __expf(m0 - bm);
      for (int c = 0; c < 64; ++c) z += __expf(ss[tid * 64 + c] - bm);
      mrow[tid] = bm; zrow[tid] = z;
    }
  }
  __syncthreads();
  if (tid < 32) {
    rowMax[rowBase + tid] = mrow[tid];
    rowInvZ[rowBase + tid] = 1.0f / zrow[tid];
  }
}

// ---------------- propagate pass: Y[32 x 512] = softmax(S) @ V, V given transposed ----------------
// 256 threads = 8 waves; waves = (row-group rg in {0,1}) x (slice/j-tile in {0..3}).
// K tile (64x512) and V^T tile (512x64) double-buffered via TDM (2 ops/block, in-order).

__global__ void __launch_bounds__(256) gcn_pass(const __bf16* __restrict__ Qb,
                                                const __bf16* __restrict__ Ktb,
                                                const __bf16* __restrict__ Vt, // [NFEAT][N_ROWS]
                                                const float* __restrict__ rowMax,
                                                const float* __restrict__ rowInvZ,
                                                float* __restrict__ Y) {       // [N_ROWS][NFEAT]
  __shared__ alignas(16) __bf16 qs[32 * NFEAT];       // 32 KiB
  __shared__ alignas(16) __bf16 kts[2][64 * NFEAT];   // 128 KiB (TDM)
  __shared__ alignas(16) __bf16 vts[2][NFEAT * 64];   // 128 KiB (TDM)
  __shared__ alignas(16) __bf16 ps[32 * 64];          // 4 KiB
  __shared__ float ms[32], zs[32];
  int tid = threadIdx.x;
  int rowBase = blockIdx.x * 32;
  int wave = tid >> 5, lane = tid & 31, hi = lane >> 4, nn = lane & 15;
  int rg = wave >> 2, sl = wave & 3;   // row-group; S j-tile == PV V-slice index

  if (wave == 0) {                     // prefetch block 0 (overlaps with qs staging)
    tdm_load_2d(lds_offset_of(kts[0]), Ktb, NFEAT, N_ROWS, NFEAT, 64, NFEAT);
    tdm_load_2d(lds_offset_of(vts[0]), Vt, N_ROWS, NFEAT, 64, NFEAT, N_ROWS);
  }
  {
    const v8bf* src = (const v8bf*)(Qb + (size_t)rowBase * NFEAT);
    v8bf* dst = (v8bf*)qs;
    for (int i = tid; i < 32 * NFEAT / 8; i += 256) dst[i] = src[i];
  }
  if (tid < 32) { ms[tid] = rowMax[rowBase + tid]; zs[tid] = rowInvZ[rowBase + tid]; }

  v8f acc[8];
#pragma unroll
  for (int t = 0; t < 8; ++t) acc[t] = (v8f){0.f,0.f,0.f,0.f,0.f,0.f,0.f,0.f};

  int p = 0;
  for (int jb = 0; jb < N_ROWS; jb += 64, p ^= 1) {
    __syncthreads();                   // buffer p^1 fully consumed; ok to overwrite
    if (wave == 0) {
      if (jb + 64 < N_ROWS) {
        tdm_load_2d(lds_offset_of(kts[p ^ 1]), Ktb + (size_t)(jb + 64) * NFEAT,
                    NFEAT, N_ROWS, NFEAT, 64, NFEAT);          // next 64 rows of K
        tdm_load_2d(lds_offset_of(vts[p ^ 1]), Vt + jb + 64,
                    N_ROWS, NFEAT, 64, NFEAT, N_ROWS);         // next 64 cols of V
        __builtin_amdgcn_s_wait_tensorcnt(2);  // current block's 2 ops done (in-order)
      } else {
        __builtin_amdgcn_s_wait_tensorcnt(0);
      }
    }
    __syncthreads();
    // S tile for (rows rg*16.., cols jb + sl*16..)
    v8f s = s_tile(qs, rg * 16, kts[p], sl * 16);
#pragma unroll
    for (int r = 0; r < 8; ++r) {
      int M = rg * 16 + r + hi * 8;
      ps[M * 64 + sl * 16 + nn] = (__bf16)__expf(s[r] - ms[M]);
    }
    __syncthreads();
    // PV: rows rg*16.., v-cols sl*128 .. +128, K-dim = this 64-col block
#pragma unroll
    for (int kk = 0; kk < 64; kk += 32) {
      v16bf a = load_frag_a(ps, 64, rg * 16, kk);
#pragma unroll
      for (int t = 0; t < 8; ++t) {
        v16bf b = load_frag_b(vts[p], 64, sl * 128 + t * 16, kk);
        acc[t] = wmma_bf16(a, b, acc[t]);
      }
    }
  }
#pragma unroll
  for (int t = 0; t < 8; ++t)
#pragma unroll
    for (int r = 0; r < 8; ++r) {
      int M = rg * 16 + r + hi * 8;
      Y[(size_t)(rowBase + M) * NFEAT + sl * 128 + t * 16 + nn] = acc[t][r] * zs[M];
    }
}

// ---------------- trailing GEMM: C[8192 x outw] = A[8192 x 512] @ W[outw x 512]^T ----------------

__global__ void __launch_bounds__(128) gcn_out(const __bf16* __restrict__ A,
                                               const __bf16* __restrict__ W,
                                               float* __restrict__ C, int outw) {
  __shared__ alignas(16) __bf16 as_[16 * NFEAT];
  int tid = threadIdx.x;
  int rowBase = blockIdx.x * 16;
  {
    const v8bf* src = (const v8bf*)(A + (size_t)rowBase * NFEAT);
    v8bf* dst = (v8bf*)as_;
    for (int i = tid; i < 16 * NFEAT / 8; i += 128) dst[i] = src[i];
  }
  __syncthreads();
  int wave = tid >> 5, lane = tid & 31, hi = lane >> 4, nn = lane & 15;
  int tpw = outw >> 6;                     // 4 (outw=256) or 2 (outw=128) tiles per wave
  v8f acc[4];
#pragma unroll
  for (int t = 0; t < 4; ++t) acc[t] = (v8f){0.f,0.f,0.f,0.f,0.f,0.f,0.f,0.f};
  for (int k = 0; k < NFEAT; k += 32) {
    v16bf a = load_frag_a(as_, NFEAT, 0, k);
    for (int t = 0; t < tpw; ++t) {
      v16bf b = load_frag_b(W, NFEAT, (wave * tpw + t) * 16, k);
      acc[t] = wmma_bf16(a, b, acc[t]);
    }
  }
  for (int t = 0; t < tpw; ++t)
#pragma unroll
    for (int r = 0; r < 8; ++r)
      C[(size_t)(rowBase + r + hi * 8) * outw + (wave * tpw + t) * 16 + nn] = acc[t][r];
}

// ---------------- host side ----------------

extern "C" void kernel_launch(void* const* d_in, const int* in_sizes, int n_in,
                              void* d_out, int out_size, void* d_ws, size_t ws_size,
                              hipStream_t stream) {
  (void)in_sizes; (void)n_in; (void)out_size; (void)ws_size;
  const float* fs = (const float*)d_in[0];   // [8192, 512]
  const float* ft = (const float*)d_in[1];   // [8192, 512]
  const float* w1 = (const float*)d_in[2];   // [256, 512]
  const float* w2 = (const float*)d_in[3];   // [128, 256]
  float* x1 = (float*)d_out;                            // [8192, 256]
  float* x2 = x1 + (size_t)N_ROWS * NHID;               // [8192, 128]

  char* ws = (char*)d_ws;
  __bf16* Qb   = (__bf16*)(ws);                               // 8 MiB  fs bf16
  __bf16* Ktb  = (__bf16*)(ws + (size_t)( 8u << 20));         // 8 MiB  ft bf16
  __bf16* fsT  = (__bf16*)(ws + (size_t)(16u << 20));         // 8 MiB  fs^T bf16
  float*  rowM = (float*) (ws + (size_t)(24u << 20));         // 32 KiB
  float*  rowZ = (float*) (ws + (size_t)(24u << 20) + 32768); // 32 KiB
  float*  Yf   = (float*) (ws + (size_t)(25u << 20));         // 16 MiB (Y1 then Y2)
  __bf16* A1b  = (__bf16*)(ws + (size_t)(41u << 20));         // 8 MiB
  __bf16* AT   = (__bf16*)(ws + (size_t)(49u << 20));         // 8 MiB  A1^T bf16
  __bf16* A2b  = (__bf16*)(ws + (size_t)(57u << 20));         // 8 MiB
  __bf16* w1b  = (__bf16*)(ws + (size_t)(65u << 20));         // 256 KiB
  __bf16* W12b = (__bf16*)(ws + (size_t)(65u << 20) + (512u << 10)); // 128 KiB

  const int nfe = N_ROWS * NFEAT;
  cvt_f32_bf16<<<(nfe + 255) / 256, 256, 0, stream>>>(fs, Qb, nfe);
  cvt_f32_bf16<<<(nfe + 255) / 256, 256, 0, stream>>>(ft, Ktb, nfe);
  transpose_f32_bf16<<<dim3(NFEAT / 32, N_ROWS / 32), dim3(32, 8), 0, stream>>>(fs, fsT, N_ROWS, NFEAT);

  gcn_stats<<<N_ROWS / 32, 256, 0, stream>>>(Qb, Ktb, rowM, rowZ);

  // A1 = softmax(S) @ fs
  gcn_pass<<<N_ROWS / 32, 256, 0, stream>>>(Qb, Ktb, fsT, rowM, rowZ, Yf);
  cvt_f32_bf16<<<(nfe + 255) / 256, 256, 0, stream>>>(Yf, A1b, nfe);
  transpose_f32_bf16<<<dim3(NFEAT / 32, N_ROWS / 32), dim3(32, 8), 0, stream>>>(Yf, AT, N_ROWS, NFEAT);

  // A2 = softmax(S) @ A1
  gcn_pass<<<N_ROWS / 32, 256, 0, stream>>>(Qb, Ktb, AT, rowM, rowZ, Yf);
  cvt_f32_bf16<<<(nfe + 255) / 256, 256, 0, stream>>>(Yf, A2b, nfe);

  cvt_f32_bf16<<<(NHID * NFEAT + 255) / 256, 256, 0, stream>>>(w1, w1b, NHID * NFEAT);
  w12_kernel<<<(NCLASS * NFEAT) / 256, 256, 0, stream>>>(w1, w2, W12b);

  // x1 = A1 @ w1^T ; x2 = A2 @ (w2@w1)^T
  gcn_out<<<N_ROWS / 16, 128, 0, stream>>>(A1b, w1b, x1, NHID);
  gcn_out<<<N_ROWS / 16, 128, 0, stream>>>(A2b, W12b, x2, NCLASS);
}
